// TemporalModule_1168231104553
// MI455X (gfx1250) — matrix operand order repaired and checked by reading, not compiled, not run on hardware
//
#include <hip/hip_runtime.h>
#include <math.h>

// ---------------------------------------------------------------------------
// Problem constants (from reference): B0 = 2*1024 rows, C=512, P=32 past,
// H=8 heads x DH=64, top-R=8, refiner hidden HID=256.
// ---------------------------------------------------------------------------
#define CC    512
#define PP    32
#define HH    8
#define DHH   64
#define RR    8
#define HID   256
#define NB0   2048
#define NWAVE 8

// padded LDS strides (elements) -> 16B aligned rows, conflict-free frags
#define PADC  520      // for 512-wide tiles   (1040 B rows)
#define PAD3C 1544     // for 1536-wide refiner input (3088 B rows)
#define PADH  264      // for 256-wide hidden tile (528 B rows)

typedef __bf16 bf16;
typedef __attribute__((ext_vector_type(16))) __bf16 v16bf;
typedef __attribute__((ext_vector_type(8)))  float  v8f;

union FragU { v16bf v; uint4 u[2]; };
union Pack8 { bf16 e[8]; uint4 u; };

// A fragment (16x32 bf16, row-major source):
// lane holds row = row0 + (lane&15); elems 0..7 = K[kA..kA+7], 8..15 = K[kA+16..kA+23],
// kA = k0 + (lane<16 ? 0 : 8)   (per CDNA5 ISA 7.12.2 16-bit A layout)
__device__ __forceinline__ v16bf frag_a(const bf16* A, int stride, int row0, int k0) {
    int lane = threadIdx.x & 31;
    int row  = row0 + (lane & 15);
    int kA   = k0 + ((lane & 16) ? 8 : 0);
    FragU f;
    f.u[0] = *(const uint4*)(A + row * stride + kA);
    f.u[1] = *(const uint4*)(A + row * stride + kA + 16);
    return f.v;
}

// B fragment (32x16 bf16) from a K-contiguous (pre-transposed) weight BT[N][K]:
// lane holds col n = n0 + (lane&15); elems 0..7 = K[kB..kB+7], 8..15 = K[kB+16..kB+23], kB = k0 + (lane<16?0:8)
__device__ __forceinline__ v16bf frag_b(const bf16* BT, int ldb, int n0, int k0) {
    int lane = threadIdx.x & 31;
    int n    = n0 + (lane & 15);
    int kB   = k0 + ((lane & 16) ? 8 : 0);
    const uint4* p = (const uint4*)(BT + n * ldb + kB);
    FragU f;
    f.u[0] = p[0];
    f.u[1] = p[2];
    return f.v;
}

__device__ __forceinline__ v8f wmma_bf16(v16bf a, v16bf b, v8f c) {
    // (neg_a, A, neg_b, B, c_mod, C, reuse_a, reuse_b)
    return __builtin_amdgcn_wmma_f32_16x16x32_bf16(false, a, false, b, (short)0, c, false, false);
}

// ---------------------------------------------------------------------------
// Prep: transpose fp32 weight [K][N] -> bf16 [N][K] so WMMA B frags are
// contiguous-in-K global loads (L2 resident, 192MB L2 holds all weights).
// ---------------------------------------------------------------------------
__global__ void transpose_to_bf16(const float* __restrict__ in, bf16* __restrict__ out,
                                  int K, int N) {
    int idx = blockIdx.x * blockDim.x + threadIdx.x;
    if (idx >= K * N) return;
    int n = idx / K, k = idx % K;     // out[n][k] = in[k][n]
    out[idx] = (bf16)in[k * N + n];
}

// ---------------------------------------------------------------------------
// Fused temporal module: 1 workgroup (256 thr / 8 waves) per batch row.
// ---------------------------------------------------------------------------
__global__ __launch_bounds__(256) void temporal_fused(
    const float* __restrict__ hidden,   // [B0, C]
    const float* __restrict__ cached,   // [B0, P, C]
    const float* __restrict__ pe,       // [32, C]
    const float* __restrict__ bo,       // [C]
    const float* __restrict__ b1,       // [HID]
    const float* __restrict__ b2,       // [2C+2]
    const bf16*  __restrict__ WqT,      // [C][C]
    const bf16*  __restrict__ WkT,
    const bf16*  __restrict__ WvT,
    const bf16*  __restrict__ WoT,
    const bf16*  __restrict__ W1T,      // [HID][3C]
    const bf16*  __restrict__ W2T,      // [2C+2][HID]
    float* __restrict__ out)            // [B0, C]
{
    // sA aliased across phases: [48][PADC] A-tile -> [16][PAD3C] refiner inp -> [16][PADC] ctx
    __shared__ __align__(16) bf16  sA[48 * PADC];
    __shared__ __align__(16) bf16  sK[33 * PADC];   // rows 0..31 k_past, row 32 k_now
    __shared__ __align__(16) bf16  sV[33 * PADC];
    __shared__ __align__(16) bf16  sH[16 * PADH];
    __shared__ __align__(16) float sQ[CC];
    __shared__ float sScore[PP];
    __shared__ int   sIdx[RR];
    __shared__ float sG[RR], sS[RR];

    const int b    = blockIdx.x;
    const int tid  = threadIdx.x;
    const int wave = tid >> 5;
    const int lane = tid & 31;

    // ---- Phase 0: A = [past+PE (32) ; now+PE (1) ; zeros (15)], vectorized 8/thr ----
    for (int i = tid; i < 48 * (CC / 8); i += 256) {
        int row = i / (CC / 8);
        int col = (i % (CC / 8)) * 8;
        Pack8 pk;
        if (row <= PP) {
            const float* src = (row < PP)
                ? (cached + (size_t)b * PP * CC + (size_t)row * CC + col)
                : (hidden + (size_t)b * CC + col);
            const float* pes = pe + (row < PP ? row : 0) * CC + col;
            float4 a0 = ((const float4*)src)[0], a1 = ((const float4*)src)[1];
            float4 p0 = ((const float4*)pes)[0], p1 = ((const float4*)pes)[1];
            pk.e[0] = (bf16)(a0.x + p0.x); pk.e[1] = (bf16)(a0.y + p0.y);
            pk.e[2] = (bf16)(a0.z + p0.z); pk.e[3] = (bf16)(a0.w + p0.w);
            pk.e[4] = (bf16)(a1.x + p1.x); pk.e[5] = (bf16)(a1.y + p1.y);
            pk.e[6] = (bf16)(a1.z + p1.z); pk.e[7] = (bf16)(a1.w + p1.w);
        } else {
            uint4 z = {0, 0, 0, 0}; pk.u = z;
        }
        *(uint4*)(sA + row * PADC + col) = pk.u;
    }
    __syncthreads();

    // ---- Phase 1a: K/V GEMMs for M-tiles 0,1 (past rows 0..31); A frags shared ----
    for (int t = wave; t < 64; t += NWAVE) {            // 2 Mtiles x 32 Ntiles
        int mt = t >> 5, nt = t & 31;
        v8f accK = {}, accV = {};
        for (int ks = 0; ks < 16; ++ks) {
            v16bf a = frag_a(sA, PADC, mt * 16, ks * 32);
            accK = wmma_bf16(a, frag_b(WkT, CC, nt * 16, ks * 32), accK);
            accV = wmma_bf16(a, frag_b(WvT, CC, nt * 16, ks * 32), accV);
        }
        int col = nt * 16 + (lane & 15);
        int rb  = mt * 16 + ((lane & 16) ? 8 : 0);      // rows <= 31 here
        #pragma unroll
        for (int j = 0; j < 8; ++j) {
            sK[(rb + j) * PADC + col] = (bf16)accK[j];
            sV[(rb + j) * PADC + col] = (bf16)accV[j];
        }
    }
    // ---- Phase 1b: M-tile 2 (row 32 = now): k_now / v_now / q_now together ----
    for (int nt = wave; nt < 32; nt += NWAVE) {
        v8f accK = {}, accV = {}, accQ = {};
        for (int ks = 0; ks < 16; ++ks) {
            v16bf a = frag_a(sA, PADC, 32, ks * 32);
            accK = wmma_bf16(a, frag_b(WkT, CC, nt * 16, ks * 32), accK);
            accV = wmma_bf16(a, frag_b(WvT, CC, nt * 16, ks * 32), accV);
            accQ = wmma_bf16(a, frag_b(WqT, CC, nt * 16, ks * 32), accQ);
        }
        if (lane < 16) {                                // D row 0 = global row 32
            int col = nt * 16 + lane;
            sK[32 * PADC + col] = (bf16)accK[0];
            sV[32 * PADC + col] = (bf16)accV[0];
            sQ[col] = accQ[0];
        }
    }
    __syncthreads();

    // ---- Phase 2: score[p] = q . k_past[p]; 8 threads per row, b128 LDS loads ----
    {
        int p = tid >> 3, sub = tid & 7;                // 32 rows x 8 threads
        float s = 0.f;
        #pragma unroll
        for (int c8 = 0; c8 < 8; ++c8) {
            int c = sub * 64 + c8 * 8;
            Pack8 kk; kk.u = *(const uint4*)(sK + p * PADC + c);
            float4 q0 = ((const float4*)(sQ + c))[0];
            float4 q1 = ((const float4*)(sQ + c))[1];
            s += q0.x * (float)kk.e[0] + q0.y * (float)kk.e[1]
               + q0.z * (float)kk.e[2] + q0.w * (float)kk.e[3]
               + q1.x * (float)kk.e[4] + q1.y * (float)kk.e[5]
               + q1.z * (float)kk.e[6] + q1.w * (float)kk.e[7];
        }
        s += __shfl_xor(s, 1, 32);
        s += __shfl_xor(s, 2, 32);
        s += __shfl_xor(s, 4, 32);
        if (sub == 0) sScore[p] = s;
    }
    __syncthreads();
    if (tid == 0) {                                     // serial top-8 over 32 (tiny)
        for (int r = 0; r < RR; ++r) {
            int best = 0; float bv = -3.0e38f;
            for (int p = 0; p < PP; ++p)
                if (sScore[p] > bv) { bv = sScore[p]; best = p; }
            sIdx[r] = best; sScore[best] = -3.0e38f;
        }
    }
    __syncthreads();

    // ---- Phase 3: refiner input [16][3C] = concat(q, k_sel, v_sel); rows 8..15 zero ----
    for (int i = tid; i < 16 * (3 * CC / 8); i += 256) {
        int row  = i / (3 * CC / 8);
        int colc = (i % (3 * CC / 8)) * 8;
        Pack8 pk;
        if (row < RR) {
            if (colc < CC) {
                float4 q0 = ((const float4*)(sQ + colc))[0];
                float4 q1 = ((const float4*)(sQ + colc))[1];
                pk.e[0] = (bf16)q0.x; pk.e[1] = (bf16)q0.y;
                pk.e[2] = (bf16)q0.z; pk.e[3] = (bf16)q0.w;
                pk.e[4] = (bf16)q1.x; pk.e[5] = (bf16)q1.y;
                pk.e[6] = (bf16)q1.z; pk.e[7] = (bf16)q1.w;
            } else if (colc < 2 * CC) {
                pk.u = *(const uint4*)(sK + sIdx[row] * PADC + (colc - CC));
            } else {
                pk.u = *(const uint4*)(sV + sIdx[row] * PADC + (colc - 2 * CC));
            }
        } else {
            uint4 z = {0, 0, 0, 0}; pk.u = z;
        }
        *(uint4*)(sA + row * PAD3C + colc) = pk.u;
    }
    __syncthreads();

    // ---- Phase 4: h = gelu(inp @ W1 + b1)  (16x256) ----
    for (int t = wave; t < 16; t += NWAVE) {
        v8f acc = {};
        for (int ks = 0; ks < 48; ++ks)
            acc = wmma_bf16(frag_a(sA, PAD3C, 0, ks * 32),
                            frag_b(W1T, 3 * CC, t * 16, ks * 32), acc);
        int col = t * 16 + (lane & 15);
        int rb  = (lane & 16) ? 8 : 0;
        #pragma unroll
        for (int j = 0; j < 8; ++j) {
            float x = acc[j] + b1[col];
            float g = 0.5f * x * (1.0f + erff(x * 0.70710678118f));   // exact gelu
            sH[(rb + j) * PADH + col] = (bf16)g;
        }
    }
    __syncthreads();

    // ---- Phase 4b: gate cols 2C (softplus) / 2C+1 (sigmoid); 16 thr per dot ----
    {
        int pair = tid >> 4, sub = tid & 15;            // 16 dots x 16 threads
        int r = pair & 7, jc = 2 * CC + (pair >> 3);
        float acc = 0.f;
        #pragma unroll
        for (int k8 = 0; k8 < 2; ++k8) {
            int k = sub * 16 + k8 * 8;
            Pack8 hh; hh.u = *(const uint4*)(sH + r * PADH + k);
            Pack8 ww; ww.u = *(const uint4*)(W2T + jc * HID + k);
            #pragma unroll
            for (int e = 0; e < 8; ++e) acc += (float)hh.e[e] * (float)ww.e[e];
        }
        acc += __shfl_xor(acc, 1, 32);
        acc += __shfl_xor(acc, 2, 32);
        acc += __shfl_xor(acc, 4, 32);
        acc += __shfl_xor(acc, 8, 32);
        if (sub == 0) {
            acc += b2[jc];
            if ((pair >> 3) == 0) {
                float sp = (acc > 20.f) ? acc : log1pf(expf(acc));
                sS[r] = sp + 1e-5f;
            } else {
                sG[r] = 1.f / (1.f + expf(-acc));
            }
        }
    }
    __syncthreads();

    // ---- Phase 5: out = h @ W2 + b2, fused gated update of k_sel/v_sel in LDS ----
    for (int t = wave; t < 64; t += NWAVE) {            // cols 0..1023 (dK | dV)
        v8f acc = {};
        for (int ks = 0; ks < 8; ++ks)
            acc = wmma_bf16(frag_a(sH, PADH, 0, ks * 32),
                            frag_b(W2T, HID, t * 16, ks * 32), acc);
        int col = t * 16 + (lane & 15);
        int rb  = (lane & 16) ? 8 : 0;
        #pragma unroll
        for (int j = 0; j < 8; ++j) {
            int r = rb + j;
            if (r < RR) {
                float delta = sG[r] * sS[r] * (acc[j] + b2[col]);
                bf16* dst = (col < CC) ? &sK[sIdx[r] * PADC + col]
                                       : &sV[sIdx[r] * PADC + (col - CC)];
                *dst = (bf16)((float)*dst + delta);
            }
        }
    }
    __syncthreads();

    // ---- Phase 6: 9-key attention, one wave per head; ctx -> sA row 0 (bf16) ----
    {
        int h  = wave;
        int d0 = h * DHH + lane * 2;
        float q0 = sQ[d0], q1 = sQ[d0 + 1];
        float logits[RR + 1];
        #pragma unroll
        for (int kk = 0; kk <= RR; ++kk) {
            int row = (kk < RR) ? sIdx[kk] : 32;
            float p = q0 * (float)sK[row * PADC + d0] + q1 * (float)sK[row * PADC + d0 + 1];
            #pragma unroll
            for (int off = 16; off > 0; off >>= 1) p += __shfl_xor(p, off, 32);
            logits[kk] = p * 0.125f;                    // DH^-0.5
        }
        float m = logits[0];
        #pragma unroll
        for (int kk = 1; kk <= RR; ++kk) m = fmaxf(m, logits[kk]);
        float w[RR + 1], denom = 0.f;
        #pragma unroll
        for (int kk = 0; kk <= RR; ++kk) { w[kk] = expf(logits[kk] - m); denom += w[kk]; }
        float inv = 1.f / denom;
        float c0 = 0.f, c1 = 0.f;
        #pragma unroll
        for (int kk = 0; kk <= RR; ++kk) {
            int row = (kk < RR) ? sIdx[kk] : 32;
            float a = w[kk] * inv;
            c0 += a * (float)sV[row * PADC + d0];
            c1 += a * (float)sV[row * PADC + d0 + 1];
        }
        sA[d0]     = (bf16)c0;                          // ctx into out-proj A row 0
        sA[d0 + 1] = (bf16)c1;
    }
    for (int i = tid; i < 15 * (CC / 8); i += 256) {    // zero rows 1..15 (b128 stores)
        int row = 1 + i / (CC / 8);
        int col = (i % (CC / 8)) * 8;
        uint4 z = {0, 0, 0, 0};
        *(uint4*)(sA + row * PADC + col) = z;
    }
    __syncthreads();

    // ---- Phase 7: out = ctx @ Wo + bo ----
    for (int t = wave; t < 32; t += NWAVE) {
        v8f acc = {};
        for (int ks = 0; ks < 16; ++ks)
            acc = wmma_bf16(frag_a(sA, PADC, 0, ks * 32),
                            frag_b(WoT, CC, t * 16, ks * 32), acc);
        if (lane < 16) {
            int col = t * 16 + lane;
            out[(size_t)b * CC + col] = acc[0] + bo[col];
        }
    }
}

// ---------------------------------------------------------------------------
extern "C" void kernel_launch(void* const* d_in, const int* in_sizes, int n_in,
                              void* d_out, int out_size, void* d_ws, size_t ws_size,
                              hipStream_t stream) {
    (void)in_sizes; (void)n_in; (void)out_size; (void)ws_size;

    const float* hidden = (const float*)d_in[0];
    const float* cached = (const float*)d_in[1];
    const float* Wq = (const float*)d_in[2];
    const float* Wk = (const float*)d_in[3];
    const float* Wv = (const float*)d_in[4];
    const float* Wo = (const float*)d_in[5];
    const float* bo = (const float*)d_in[6];
    const float* W1 = (const float*)d_in[7];
    const float* b1 = (const float*)d_in[8];
    const float* W2 = (const float*)d_in[9];
    const float* b2 = (const float*)d_in[10];
    const float* pe = (const float*)d_in[11];
    // d_in[12] = select_top_r == 8 (compile-time RR)

    char* ws   = (char*)d_ws;
    bf16* WqT  = (bf16*)(ws);
    bf16* WkT  = (bf16*)(ws + 1 * 524288);
    bf16* WvT  = (bf16*)(ws + 2 * 524288);
    bf16* WoT  = (bf16*)(ws + 3 * 524288);
    bf16* W1T  = (bf16*)(ws + 4 * 524288);            // [256][1536] bf16
    bf16* W2T  = (bf16*)(ws + 4 * 524288 + 786432);   // [1026][256] bf16

    const int thr = 256;
    transpose_to_bf16<<<(512 * 512 + thr - 1) / thr, thr, 0, stream>>>(Wq, WqT, 512, 512);
    transpose_to_bf16<<<(512 * 512 + thr - 1) / thr, thr, 0, stream>>>(Wk, WkT, 512, 512);
    transpose_to_bf16<<<(512 * 512 + thr - 1) / thr, thr, 0, stream>>>(Wv, WvT, 512, 512);
    transpose_to_bf16<<<(512 * 512 + thr - 1) / thr, thr, 0, stream>>>(Wo, WoT, 512, 512);
    transpose_to_bf16<<<(1536 * 256 + thr - 1) / thr, thr, 0, stream>>>(W1, W1T, 1536, 256);
    transpose_to_bf16<<<(256 * 1026 + thr - 1) / thr, thr, 0, stream>>>(W2, W2T, 256, 1026);

    temporal_fused<<<NB0, 256, 0, stream>>>(hidden, cached, pe, bo, b1, b2,
                                            WqT, WkT, WvT, WoT, W1T, W2T,
                                            (float*)d_out);
}
